// A8W4LlamaMLP_81913616270140
// MI455X (gfx1250) — 8-bit matrix operand order repaired and checked by reading, compiled, not run on hardware
//
#include <hip/hip_runtime.h>
#include <stdint.h>


typedef __attribute__((ext_vector_type(8))) int v8i;

#define LDS_STRIDE 80
#define KSTEP 64

__device__ __forceinline__ void async_copy_b128(unsigned lds_addr,
                                                const void* gbase,
                                                unsigned goff) {
    asm volatile("global_load_async_to_lds_b128 %0, %1, %2"
                 :: "v"(lds_addr), "v"(goff),
                    "s"((unsigned long long)(uintptr_t)gbase)
                 : "memory");
}
__device__ __forceinline__ void async_wait0() {
    asm volatile("s_wait_asynccnt 0x0" ::: "memory");
}
__device__ __forceinline__ unsigned lds_off(const void* p) {
    return (unsigned)(uintptr_t)p;
}

__device__ __forceinline__ v8i load_frag_a(const char* tile) {
    const int lane = threadIdx.x & 31;
    const int row  = lane & 15;
    const int kb   = (lane >> 4) * 16;
    const char* p = tile + row * LDS_STRIDE + kb;
    union { v8i v; unsigned long long q[4]; } f;
    f.q[0] = *(const unsigned long long*)(p +  0);
    f.q[1] = *(const unsigned long long*)(p +  8);
    f.q[2] = *(const unsigned long long*)(p + 32);
    f.q[3] = *(const unsigned long long*)(p + 40);
    return f.v;
}
__device__ __forceinline__ v8i load_frag_b(const char* tile) {
    const int lane = threadIdx.x & 31;
    const int n  = lane & 15;
    const int kb = (lane >> 4) * 16;
    const char* p = tile + n * LDS_STRIDE + kb;
    union { v8i v; int4 o[2]; } f;
    f.o[0] = *(const int4*)(p +  0);
    f.o[1] = *(const int4*)(p + 32);
    return f.v;
}

__device__ __forceinline__ int pack4dw(int4 a) {
    unsigned t01 = __builtin_amdgcn_perm((unsigned)a.y, (unsigned)a.x, 0x00000400u);
    unsigned t23 = __builtin_amdgcn_perm((unsigned)a.w, (unsigned)a.z, 0x00000400u);
    return (int)__builtin_amdgcn_perm(t23, t01, 0x05040100u);
}

__global__ __launch_bounds__(256) void pack_i32_to_i8_kernel(
    const int* __restrict__ src, int8_t* __restrict__ dst, int n16) {
    const int i = blockIdx.x * 256 + threadIdx.x;
    if (i >= n16) return;
    const int4* p = (const int4*)src + (long long)i * 4;
    int4 w;
    w.x = pack4dw(p[0]);
    w.y = pack4dw(p[1]);
    w.z = pack4dw(p[2]);
    w.w = pack4dw(p[3]);
    *(int4*)(dst + (long long)i * 16) = w;
}

__global__ __launch_bounds__(256) void a8w4_gateup_kernel(
    const int8_t* __restrict__ xp,
    const int8_t* __restrict__ gwp,
    const int8_t* __restrict__ uwp,
    const float* __restrict__ ga, const float* __restrict__ gb,
    const float* __restrict__ ua, const float* __restrict__ ub,
    const float* __restrict__ dscale,
    int8_t* __restrict__ hq,
    int H, int I)
{
    constexpr unsigned ABUF = 64 * LDS_STRIDE;
    constexpr unsigned BBUF = 128 * LDS_STRIDE;
    __shared__ __align__(16) char lA [2 * 64  * LDS_STRIDE];
    __shared__ __align__(16) char lBg[2 * 128 * LDS_STRIDE];
    __shared__ __align__(16) char lBu[2 * 128 * LDS_STRIDE];

    const int t     = threadIdx.x;
    const int wave  = t >> 5;
    const int waveM = wave >> 2;
    const int waveN = wave & 3;
    const int mBase = blockIdx.x * 64;
    const int nBase = blockIdx.y * 128;

    const v8i vzero = {0, 0, 0, 0, 0, 0, 0, 0};
    v8i accg[2][2], accu[2][2];
#pragma unroll
    for (int i = 0; i < 2; ++i)
#pragma unroll
        for (int j = 0; j < 2; ++j) { accg[i][j] = vzero; accu[i][j] = vzero; }

    const int srow = t >> 2, soff = (t & 3) * 16;
    const unsigned ldsA  = lds_off(lA)  + srow * LDS_STRIDE + soff;
    const unsigned ldsBg = lds_off(lBg) + srow * LDS_STRIDE + soff;
    const unsigned ldsBu = lds_off(lBu) + srow * LDS_STRIDE + soff;
    const unsigned gA  = (unsigned)((mBase + srow) * H + soff);
    const unsigned gB0 = (unsigned)((nBase + srow) * H + soff);
    const unsigned gB1 = gB0 + (unsigned)(64 * H);

    auto stage = [&](unsigned k0, int b) {
        const unsigned lb = (unsigned)b;
        async_copy_b128(ldsA  + lb * ABUF,                    xp,  gA  + k0);
        async_copy_b128(ldsBg + lb * BBUF,                    gwp, gB0 + k0);
        async_copy_b128(ldsBg + lb * BBUF + 64 * LDS_STRIDE,  gwp, gB1 + k0);
        async_copy_b128(ldsBu + lb * BBUF,                    uwp, gB0 + k0);
        async_copy_b128(ldsBu + lb * BBUF + 64 * LDS_STRIDE,  uwp, gB1 + k0);
    };

    const int KS = H / KSTEP;
    stage(0u, 0);
    async_wait0();
    __syncthreads();

    for (int ks = 0; ks < KS; ++ks) {
        if (ks + 1 < KS) stage((unsigned)((ks + 1) * KSTEP), (ks + 1) & 1);

        const char* bufA  = lA  + (ks & 1) * ABUF;
        const char* bufBg = lBg + (ks & 1) * BBUF;
        const char* bufBu = lBu + (ks & 1) * BBUF;

        v8i a[2];
#pragma unroll
        for (int mt = 0; mt < 2; ++mt)
            a[mt] = load_frag_a(bufA + (waveM * 32 + mt * 16) * LDS_STRIDE);
#pragma unroll
        for (int nt = 0; nt < 2; ++nt) {
            const v8i bg = load_frag_b(bufBg + (waveN * 32 + nt * 16) * LDS_STRIDE);
            const v8i bu = load_frag_b(bufBu + (waveN * 32 + nt * 16) * LDS_STRIDE);
#pragma unroll
            for (int mt = 0; mt < 2; ++mt) {
                accg[mt][nt] = __builtin_amdgcn_wmma_i32_16x16x64_iu8(
                    true, a[mt], true, bg, accg[mt][nt], false, false);
                accu[mt][nt] = __builtin_amdgcn_wmma_i32_16x16x64_iu8(
                    true, a[mt], true, bu, accu[mt][nt], false, false);
            }
        }
        async_wait0();
        __syncthreads();
    }

    const int lane = t & 31;
    const float inv_s = 1.0f / dscale[0];
#pragma unroll
    for (int nt = 0; nt < 2; ++nt) {
        const int col = nBase + waveN * 32 + nt * 16 + (lane & 15);
        const float agv = ga[col], gbv = gb[col];
        const float auv = ua[col], ubv = ub[col];
#pragma unroll
        for (int mt = 0; mt < 2; ++mt) {
            const int mrow0 = mBase + waveM * 32 + mt * 16 + ((lane >> 4) * 8);
#pragma unroll
            for (int d = 0; d < 8; ++d) {
                const float g = (float)accg[mt][nt][d] * agv + gbv;
                const float u = (float)accu[mt][nt][d] * auv + ubv;
                const float h = (g / (1.0f + __expf(-g))) * u;
                float q = rintf(h * inv_s);
                q = fminf(127.0f, fmaxf(-128.0f, q));
                hq[(long long)(mrow0 + d) * I + col] = (int8_t)(int)q;
            }
        }
    }
}

__global__ __launch_bounds__(256) void a8w4_down_kernel(
    const int8_t* __restrict__ hq,
    const int8_t* __restrict__ dwp,
    const float* __restrict__ da, const float* __restrict__ db,
    float* __restrict__ out,
    int I, int H)
{
    constexpr unsigned ABUF = 64 * LDS_STRIDE;
    constexpr unsigned BBUF = 128 * LDS_STRIDE;
    __shared__ __align__(16) char lA[2 * 64  * LDS_STRIDE];
    __shared__ __align__(16) char lB[2 * 128 * LDS_STRIDE];

    const int t     = threadIdx.x;
    const int wave  = t >> 5;
    const int waveM = wave >> 2;
    const int waveN = wave & 3;
    const int mBase = blockIdx.x * 64;
    const int nBase = blockIdx.y * 128;

    const v8i vzero = {0, 0, 0, 0, 0, 0, 0, 0};
    v8i acc[2][2];
#pragma unroll
    for (int i = 0; i < 2; ++i)
#pragma unroll
        for (int j = 0; j < 2; ++j) acc[i][j] = vzero;

    const int srow = t >> 2, soff = (t & 3) * 16;
    const unsigned ldsA = lds_off(lA) + srow * LDS_STRIDE + soff;
    const unsigned ldsB = lds_off(lB) + srow * LDS_STRIDE + soff;
    const unsigned gA  = (unsigned)((mBase + srow) * I + soff);
    const unsigned gB0 = (unsigned)((nBase + srow) * I + soff);
    const unsigned gB1 = gB0 + (unsigned)(64 * I);

    auto stage = [&](unsigned k0, int b) {
        const unsigned lb = (unsigned)b;
        async_copy_b128(ldsA + lb * ABUF,                   hq,  gA  + k0);
        async_copy_b128(ldsB + lb * BBUF,                   dwp, gB0 + k0);
        async_copy_b128(ldsB + lb * BBUF + 64 * LDS_STRIDE, dwp, gB1 + k0);
    };

    const int KS = I / KSTEP;
    stage(0u, 0);
    async_wait0();
    __syncthreads();

    for (int ks = 0; ks < KS; ++ks) {
        if (ks + 1 < KS) stage((unsigned)((ks + 1) * KSTEP), (ks + 1) & 1);

        const char* bufA = lA + (ks & 1) * ABUF;
        const char* bufB = lB + (ks & 1) * BBUF;

        v8i a[2];
#pragma unroll
        for (int mt = 0; mt < 2; ++mt)
            a[mt] = load_frag_a(bufA + (waveM * 32 + mt * 16) * LDS_STRIDE);
#pragma unroll
        for (int nt = 0; nt < 2; ++nt) {
            const v8i b = load_frag_b(bufB + (waveN * 32 + nt * 16) * LDS_STRIDE);
#pragma unroll
            for (int mt = 0; mt < 2; ++mt) {
                acc[mt][nt] = __builtin_amdgcn_wmma_i32_16x16x64_iu8(
                    true, a[mt], true, b, acc[mt][nt], false, false);
            }
        }
        async_wait0();
        __syncthreads();
    }

    const int lane = t & 31;
#pragma unroll
    for (int nt = 0; nt < 2; ++nt) {
        const int col = nBase + waveN * 32 + nt * 16 + (lane & 15);
        const float av = da[col], bv = db[col];
#pragma unroll
        for (int mt = 0; mt < 2; ++mt) {
            const int mrow0 = mBase + waveM * 32 + mt * 16 + ((lane >> 4) * 8);
#pragma unroll
            for (int d = 0; d < 8; ++d) {
                out[(long long)(mrow0 + d) * H + col] =
                    (float)acc[mt][nt][d] * av + bv;
            }
        }
    }
}

extern "C" void kernel_launch(void* const* d_in, const int* in_sizes, int n_in,
                              void* d_out, int out_size, void* d_ws, size_t ws_size,
                              hipStream_t stream) {
    (void)in_sizes; (void)n_in; (void)out_size; (void)ws_size;
    const int T = 4096, H = 4096, I = 11008;

    const int*   x   = (const int*)d_in[0];
    const int*   gw  = (const int*)d_in[1];
    const int*   uw  = (const int*)d_in[2];
    const int*   dw  = (const int*)d_in[3];
    const float* ga  = (const float*)d_in[4];
    const float* gb  = (const float*)d_in[5];
    const float* ua  = (const float*)d_in[6];
    const float* ub  = (const float*)d_in[7];
    const float* da  = (const float*)d_in[8];
    const float* db  = (const float*)d_in[9];
    const float* dsc = (const float*)d_in[10];

    char* ws = (char*)d_ws;
    const long long HQ_SZ = (long long)T * I;
    const long long XP_SZ = (long long)T * H;
    const long long W_SZ  = (long long)I * H;
    int8_t* hq  = (int8_t*)(ws);
    int8_t* xp  = (int8_t*)(ws + HQ_SZ);
    int8_t* gwp = (int8_t*)(ws + HQ_SZ + XP_SZ);
    int8_t* uwp = (int8_t*)(ws + HQ_SZ + XP_SZ + W_SZ);
    int8_t* dwp = gwp;
    float*  out = (float*)d_out;

    dim3 blk(256, 1, 1);
    const int nx16 = (int)(XP_SZ / 16);
    const int nw16 = (int)(W_SZ / 16);

    pack_i32_to_i8_kernel<<<(nx16 + 255) / 256, blk, 0, stream>>>(x,  xp,  nx16);
    pack_i32_to_i8_kernel<<<(nw16 + 255) / 256, blk, 0, stream>>>(gw, gwp, nw16);
    pack_i32_to_i8_kernel<<<(nw16 + 255) / 256, blk, 0, stream>>>(uw, uwp, nw16);

    a8w4_gateup_kernel<<<dim3(T / 64, I / 128), blk, 0, stream>>>(
        xp, gwp, uwp, ga, gb, ua, ub, dsc, hq, H, I);

    pack_i32_to_i8_kernel<<<(nw16 + 255) / 256, blk, 0, stream>>>(dw, dwp, nw16);
    a8w4_down_kernel<<<dim3(T / 64, H / 128), blk, 0, stream>>>(
        hq, dwp, da, db, out, I, H);
}
